// AnomalyTransformer_51840255263101
// MI455X (gfx1250) — matrix operand order repaired and checked by reading, not compiled, not run on hardware
//
#include <hip/hip_runtime.h>
#include <math.h>

// ---------------- problem constants ----------------
#define L_WIN   512
#define CIN     55
#define DMODEL  512
#define NH      8
#define DHEAD   64
#define NB      4
#define MROWS   (NB * L_WIN)          // 2048
#define E_LAYERS 3

typedef __attribute__((ext_vector_type(16))) _Float16 v16h;
typedef __attribute__((ext_vector_type(8)))  _Float16 h8;
typedef __attribute__((ext_vector_type(8)))  float    v8f;

enum { ACT_NONE = 0, ACT_RELU = 1, ACT_GELU = 2 };

// ---------------- generic f16-WMMA GEMM ----------------
// C[M,N] = act( cscale * (A[M,K] @ B[K,N]) + bias[N] + tab[(m%tabmod)*ldtab+n] )
// bt=1 means B(k,n) is read from Bw[n*ldb + k] (i.e. B given transposed).
// Batched over blockIdx.z: z = z1*batchH + z2, pointer offsets z1*s?1 + z2*s?2.
#define KC   128
#define LDSS (KC + 8)     // +8 halves => row stride 272B, conflict-free b128 reads

__global__ __launch_bounds__(256) void gemm_f16_wmma(
    const float* __restrict__ A, const float* __restrict__ Bw,
    const float* __restrict__ bias, const float* __restrict__ tab,
    float* __restrict__ C,
    int M, int N, int K, int lda, int ldb, int ldc,
    int bt, float cscale, int act, int tabmod, int ldtab,
    int batchH,
    long sA1, long sA2, long sB1, long sB2, long sC1, long sC2)
{
  __shared__ __align__(16) _Float16 As[64 * LDSS];
  __shared__ __align__(16) _Float16 Bs[64 * LDSS];

  const int z  = blockIdx.z;
  const int z1 = z / batchH, z2 = z - z1 * batchH;
  const float* Ab = A  + z1 * sA1 + z2 * sA2;
  const float* Bb = Bw + z1 * sB1 + z2 * sB2;
  float*       Cb = C  + z1 * sC1 + z2 * sC2;

  const int m0 = blockIdx.y * 64;
  const int n0 = blockIdx.x * 64;
  const int tid  = threadIdx.x;
  const int lane = tid & 31;
  const int wave = tid >> 5;
  const int rm  = (wave >> 1) * 16;      // wave row base (4 row-tiles)
  const int cn0 = (wave & 1) * 32;       // wave col base (2 col-tiles of 16)
  const int cn1 = cn0 + 16;
  const int lm = lane & 15;
  const int lg = lane >> 4;

  v8f acc0 = {};
  v8f acc1 = {};

  for (int kp = 0; kp < K; kp += KC) {
    const int krem = K - kp;
    const int kc = (krem < KC) ? ((krem + 31) & ~31) : KC;

    // A tile: row-major 64 x KC (zero padded)
    for (int idx = tid; idx < 64 * KC; idx += 256) {
      const int r = idx >> 7, k = idx & (KC - 1);
      const int gm = m0 + r, gk = kp + k;
      float v = 0.f;
      if (gm < M && gk < K) v = Ab[(long)gm * lda + gk];
      As[r * LDSS + k] = (_Float16)v;
    }
    // B tile: stored n-major (Bs[n][k]) so fragment reads are contiguous
    for (int idx = tid; idx < 64 * KC; idx += 256) {
      const int n = idx >> 7, k = idx & (KC - 1);
      const int gn = n0 + n, gk = kp + k;
      float v = 0.f;
      if (gn < N && gk < K)
        v = bt ? Bb[(long)gn * ldb + gk] : Bb[(long)gk * ldb + gn];
      Bs[n * LDSS + k] = (_Float16)v;
    }
    __syncthreads();

    for (int kk = 0; kk < kc; kk += 32) {
      // A fragment (16x32 f16): lane lm+16*lg holds K = lg*16 + i (contiguous)
      const _Float16* ap = &As[(rm + lm) * LDSS + kk + lg * 16];
      h8 alo = *(const h8*)(ap);
      h8 ahi = *(const h8*)(ap + 8);
      v16h af;
#pragma unroll
      for (int i = 0; i < 8; ++i) { af[i] = alo[i]; af[8 + i] = ahi[i]; }

      // B fragments (32x16 f16): lane holds N=lm, K = lg*16 + i (contiguous)
      const _Float16* bp0 = &Bs[(cn0 + lm) * LDSS + kk + lg * 16];
      h8 b0l = *(const h8*)(bp0);
      h8 b0h = *(const h8*)(bp0 + 8);
      v16h bf0;
#pragma unroll
      for (int i = 0; i < 8; ++i) { bf0[i] = b0l[i]; bf0[8 + i] = b0h[i]; }

      const _Float16* bp1 = &Bs[(cn1 + lm) * LDSS + kk + lg * 16];
      h8 b1l = *(const h8*)(bp1);
      h8 b1h = *(const h8*)(bp1 + 8);
      v16h bf1;
#pragma unroll
      for (int i = 0; i < 8; ++i) { bf1[i] = b1l[i]; bf1[8 + i] = b1h[i]; }

      acc0 = __builtin_amdgcn_wmma_f32_16x16x32_f16(false, af, false, bf0,
                                                    (short)0, acc0, false, false);
      acc1 = __builtin_amdgcn_wmma_f32_16x16x32_f16(false, af, false, bf1,
                                                    (short)0, acc1, false, false);
    }
    __syncthreads();
  }

  // epilogue: C/D layout -> VGPR j: M = rm + j + 8*lg, N = tile + lm
#pragma unroll
  for (int t = 0; t < 2; ++t) {
    v8f acc = t ? acc1 : acc0;
    const int cb = t ? cn1 : cn0;
    const int gn = n0 + cb + lm;
    if (gn < N) {
      const float bv = bias ? bias[gn] : 0.f;
#pragma unroll
      for (int j = 0; j < 8; ++j) {
        const int gm = m0 + rm + j + 8 * lg;
        if (gm < M) {
          float v = acc[j] * cscale + bv;
          if (tab) v += tab[(long)(gm % tabmod) * ldtab + gn];
          if (act == ACT_RELU)      v = fmaxf(v, 0.f);
          else if (act == ACT_GELU) v = 0.5f * v * (1.f + erff(v * 0.70710678f));
          Cb[(long)gm * ldc + gn] = v;
        }
      }
    }
  }
}

// ---------------- front-end: stats + graph MLP -> alpha, res ----------------
__global__ __launch_bounds__(64) void frontend_kernel(
    const float* __restrict__ x, const float* __restrict__ ne,
    const float* __restrict__ lfc1w, const float* __restrict__ lfc1b,
    const float* __restrict__ lfc2w, const float* __restrict__ lfc2b,
    const float* __restrict__ gfcw,  const float* __restrict__ gfcb,
    const float* __restrict__ gqw,   const float* __restrict__ gqb,
    const float* __restrict__ gkw,   const float* __restrict__ gkb,
    const float* __restrict__ ffcw,  const float* __restrict__ ffcb,
    const float* __restrict__ ofcw,  const float* __restrict__ ofcb,
    const float* __restrict__ hng,   const float* __restrict__ hnb,
    float* __restrict__ res, float* __restrict__ alpha)
{
  __shared__ float qs[CIN][16], ks2[CIN][16], hgs[CIN][16], hls[CIN][16];
  const int b = blockIdx.x, c = threadIdx.x;
  const float* xb = x + (long)b * L_WIN * CIN;
  if (c < CIN) {
    float sa = 0.f, mx = 0.f, s2 = 0.f;
    for (int l = 0; l < L_WIN; ++l) {
      float s = 0.f;
      for (int t = -2; t <= 2; ++t) {
        int u = l + t; u = u < 0 ? 0 : (u > L_WIN - 1 ? L_WIN - 1 : u);
        s += xb[u * CIN + c];
      }
      const float r = xb[l * CIN + c] - s * 0.2f;
      res[((long)b * L_WIN + l) * CIN + c] = r;
      const float a = fabsf(r);
      sa += a; mx = fmaxf(mx, a); s2 += r * r;
    }
    const float ma = sa * (1.f / L_WIN), m2 = s2 * (1.f / L_WIN);
    const float sd = sqrtf(fmaxf(m2 - ma * ma, 0.f));
    const float h0[4] = {ma, mx, sd, m2};
    float hid[16];
    for (int j = 0; j < 16; ++j) {
      float v = lfc1b[j];
      for (int i = 0; i < 4; ++i) v += h0[i] * lfc1w[i * 16 + j];
      hid[j] = fmaxf(v, 0.f);
    }
    for (int j = 0; j < 16; ++j) {
      float v = lfc2b[j];
      for (int i = 0; i < 16; ++i) v += hid[i] * lfc2w[i * 16 + j];
      hls[c][j] = v;
    }
    for (int j = 0; j < 16; ++j) {
      float vq = gqb[j], vk = gkb[j], vg = gfcb[j];
      for (int i = 0; i < 4; ++i) {
        vq += h0[i] * gqw[i * 16 + j];
        vk += h0[i] * gkw[i * 16 + j];
        vg += h0[i] * gfcw[i * 16 + j];
      }
      qs[c][j] = vq; ks2[c][j] = vk; hgs[c][j] = vg;
    }
  }
  __syncthreads();
  if (c < CIN) {
    float srow[CIN]; float mxs = -3.4e38f;
    for (int d = 0; d < CIN; ++d) {
      float sq = 0.f, sp = 0.f;
      for (int j = 0; j < 16; ++j) sq += qs[c][j] * ks2[d][j];
      for (int j = 0; j < 16; ++j) sp += ne[c * 16 + j] * ne[d * 16 + j];
      const float v = fmaxf(sq, 0.f) + 0.2f * fmaxf(sp, 0.f) + (c == d ? 1.f : 0.f);
      srow[d] = v; mxs = fmaxf(mxs, v);
    }
    float ssum = 0.f;
    for (int d = 0; d < CIN; ++d) { srow[d] = expf(srow[d] - mxs); ssum += srow[d]; }
    const float inv = 1.f / ssum;
    float agg[16];
    for (int j = 0; j < 16; ++j) agg[j] = 0.f;
    for (int d = 0; d < CIN; ++d) {
      const float w = srow[d] * inv;
      for (int j = 0; j < 16; ++j) agg[j] += w * hgs[d][j];
    }
    float fu[16]; float mu = 0.f;
    for (int j = 0; j < 16; ++j) {
      const float hgr = hgs[c][j] + 0.3f * fmaxf(agg[j], 0.f);
      const float f = hls[c][j] + 0.1f * hgr;
      fu[j] = f; mu += f;
    }
    mu *= (1.f / 16.f);
    float var = 0.f;
    for (int j = 0; j < 16; ++j) { const float d2 = fu[j] - mu; var += d2 * d2; }
    var *= (1.f / 16.f);
    const float rs = rsqrtf(var + 1e-5f);
    float lnv[16];
    for (int j = 0; j < 16; ++j) lnv[j] = (fu[j] - mu) * rs * hng[j] + hnb[j];
    float o = ofcb[0];
    for (int j = 0; j < 16; ++j) {
      float v = ffcb[j];
      for (int i = 0; i < 16; ++i) v += lnv[i] * ffcw[i * 16 + j];
      o += fmaxf(v, 0.f) * ofcw[j];
    }
    alpha[b * CIN + c] = 1.2f + 0.3f * tanhf(o);
  }
}

// x_enh = x + alpha[b,c] * res
__global__ void enhance_kernel(const float* __restrict__ x, const float* __restrict__ res,
                               const float* __restrict__ alpha, float* __restrict__ xe)
{
  const int i = blockIdx.x * blockDim.x + threadIdx.x;
  if (i >= NB * L_WIN * CIN) return;
  const int c = i % CIN;
  const int b = i / (L_WIN * CIN);
  xe[i] = x[i] + alpha[b * CIN + c] * res[i];
}

// sinusoidal PE table [L, D]
__global__ void pos_emb_kernel(float* __restrict__ pe)
{
  const int i = blockIdx.x * blockDim.x + threadIdx.x;
  if (i >= L_WIN * (DMODEL / 2)) return;
  const int l = i / (DMODEL / 2), j = i - l * (DMODEL / 2);
  const float div = expf((float)(2 * j) * (-9.210340371976184f / DMODEL));
  const float a = (float)l * div;
  pe[l * DMODEL + 2 * j]     = sinf(a);
  pe[l * DMODEL + 2 * j + 1] = cosf(a);
}

// gather circular conv-1d patches: Ae[m, t*55+c] = xe[b, (l-1+t) mod L, c]
__global__ void gather_emb_kernel(const float* __restrict__ xe, float* __restrict__ Ae)
{
  const int i = blockIdx.x * blockDim.x + threadIdx.x;
  if (i >= MROWS * 165) return;
  const int m = i / 165, j = i - m * 165;
  const int t = j / CIN, c = j - t * CIN;
  const int b = m >> 9, l = m & 511;
  const int row = (l - 1 + t + L_WIN) & 511;
  Ae[i] = xe[((long)(b << 9) + row) * CIN + c];
}

// sigma heads: sig[b,h,l] = 3^(sigmoid(5*(x@wsig+b))+1e-5) - 1
__global__ void sigma_head_kernel(const float* __restrict__ x,
                                  const float* __restrict__ wsig, const float* __restrict__ bsig,
                                  float* __restrict__ sig)
{
  const int t = blockIdx.x * blockDim.x + threadIdx.x;
  if (t >= MROWS * NH) return;
  const int m = t >> 3, h = t & 7;
  const float* row = x + (long)m * DMODEL;
  float acc = bsig[h];
  for (int d = 0; d < DMODEL; ++d) acc += row[d] * wsig[d * NH + h];
  float sg = 1.f / (1.f + expf(-5.f * acc)) + 1e-5f;
  sg = expf(sg * 1.0986122886681098f) - 1.f;   // 3^sg - 1
  const int b = m >> 9, l = m & 511;
  sig[((b * NH) + h) * L_WIN + l] = sg;
}

// row softmax over 512 cols, one wave32 per row, in place
__global__ __launch_bounds__(256) void softmax_rows(float* __restrict__ p, int rows)
{
  const int wave = threadIdx.x >> 5, lane = threadIdx.x & 31;
  const int row = blockIdx.x * 8 + wave;
  if (row >= rows) return;
  float* r = p + (long)row * L_WIN;
  float v[16]; float mx = -3.4e38f;
#pragma unroll
  for (int i = 0; i < 16; ++i) { v[i] = r[lane + i * 32]; mx = fmaxf(mx, v[i]); }
#pragma unroll
  for (int o = 16; o > 0; o >>= 1) mx = fmaxf(mx, __shfl_xor(mx, o, 32));
  float s = 0.f;
#pragma unroll
  for (int i = 0; i < 16; ++i) { v[i] = expf(v[i] - mx); s += v[i]; }
#pragma unroll
  for (int o = 16; o > 0; o >>= 1) s += __shfl_xor(s, o, 32);
  const float inv = 1.f / s;
#pragma unroll
  for (int i = 0; i < 16; ++i) r[lane + i * 32] = v[i] * inv;
}

// fused prior & sigma broadcast: one pass, one write each
__global__ void prior_sigma_kernel(const float* __restrict__ sig,
                                   float* __restrict__ prior, float* __restrict__ sigma_out,
                                   long n)
{
  const long i = (long)blockIdx.x * blockDim.x + threadIdx.x;
  if (i >= n) return;
  const long r = i >> 9;
  const int s = (int)(i & 511);
  const int l = (int)(r & 511);
  const float sg = sig[r];
  const float d = (float)(l - s);
  const float pr = expf(-(d * d) / (2.f * sg * sg)) * 0.3989422804014327f / sg;
  prior[i] = pr;
  sigma_out[i] = sg;
}

// dst = LN(a (+ b)) * g + be, wave32 per row, D=512
__global__ __launch_bounds__(256) void ln_residual(
    const float* __restrict__ a, const float* __restrict__ b,
    const float* __restrict__ g, const float* __restrict__ be,
    float* __restrict__ dst, int Mrows)
{
  const int wave = threadIdx.x >> 5, lane = threadIdx.x & 31;
  const int row = blockIdx.x * 8 + wave;
  if (row >= Mrows) return;
  const float* pa = a + (long)row * DMODEL;
  const float* pb = b ? b + (long)row * DMODEL : nullptr;
  float vals[16]; float s1 = 0.f, s2 = 0.f;
#pragma unroll
  for (int i = 0; i < 16; ++i) {
    const int c = lane + i * 32;
    float v = pa[c];
    if (pb) v += pb[c];
    vals[i] = v; s1 += v; s2 += v * v;
  }
#pragma unroll
  for (int o = 16; o > 0; o >>= 1) { s1 += __shfl_xor(s1, o, 32); s2 += __shfl_xor(s2, o, 32); }
  const float mu = s1 * (1.f / DMODEL);
  const float var = s2 * (1.f / DMODEL) - mu * mu;
  const float rs = rsqrtf(var + 1e-5f);
  float* pd = dst + (long)row * DMODEL;
#pragma unroll
  for (int i = 0; i < 16; ++i) {
    const int c = lane + i * 32;
    pd[c] = (vals[i] - mu) * rs * g[c] + be[c];
  }
}

// ---------------- host-side launch ----------------
static inline void gemm(hipStream_t st, const float* A, const float* B, const float* bias,
                        const float* tab, float* C, int M, int N, int K,
                        int lda, int ldb, int ldc, int bt, float cscale, int act,
                        int tabmod, int ldtab,
                        int Z = 1, int batchH = 1,
                        long sA1 = 0, long sA2 = 0, long sB1 = 0, long sB2 = 0,
                        long sC1 = 0, long sC2 = 0)
{
  dim3 grid((N + 63) / 64, (M + 63) / 64, Z);
  gemm_f16_wmma<<<grid, 256, 0, st>>>(A, B, bias, tab, C, M, N, K, lda, ldb, ldc,
                                      bt, cscale, act, tabmod, ldtab, batchH,
                                      sA1, sA2, sB1, sB2, sC1, sC2);
}

extern "C" void kernel_launch(void* const* d_in, const int* in_sizes, int n_in,
                              void* d_out, int out_size, void* d_ws, size_t ws_size,
                              hipStream_t stream)
{
  // input order: x, then params in dict insertion order, then 3 layer dicts (18 each)
  const float* X  = (const float*)d_in[0];
  const float* NE = (const float*)d_in[1];
  const float* TOKW  = (const float*)d_in[18];   // [3,55,512] == [165,512] row-major
  const float* LNFG  = (const float*)d_in[19];
  const float* LNFB  = (const float*)d_in[20];
  const float* PROJW = (const float*)d_in[21];
  const float* PROJB = (const float*)d_in[22];

  float* ws = (float*)d_ws;
  size_t o = 0;
  float* RES = ws + o; o += 112640;
  float* XE  = ws + o; o += 112640;
  float* AL  = ws + o; o += 256;
  float* PE  = ws + o; o += 262144;
  float* AE  = ws + o; o += 337920;
  float* OUT = ws + o; o += 1048576;
  float* Q   = ws + o; o += 1048576;
  float* Kb  = ws + o; o += 1048576;
  float* V   = ws + o; o += 1048576;
  float* ATT = ws + o; o += 1048576;
  float* TMP = ws + o; o += 1048576;
  float* XR  = ws + o; o += 1048576;
  float* FF  = ws + o; o += 1048576;
  float* SIG = ws + o; o += 16384;

  // ---- front end ----
  frontend_kernel<<<NB, 64, 0, stream>>>(
      X, NE,
      (const float*)d_in[2],  (const float*)d_in[3],   // lfc1
      (const float*)d_in[4],  (const float*)d_in[5],   // lfc2
      (const float*)d_in[6],  (const float*)d_in[7],   // gfc
      (const float*)d_in[8],  (const float*)d_in[9],   // gq
      (const float*)d_in[10], (const float*)d_in[11],  // gk
      (const float*)d_in[12], (const float*)d_in[13],  // ffc
      (const float*)d_in[14], (const float*)d_in[15],  // ofc
      (const float*)d_in[16], (const float*)d_in[17],  // hn
      RES, AL);
  enhance_kernel<<<(NB * L_WIN * CIN + 255) / 256, 256, 0, stream>>>(X, RES, AL, XE);
  pos_emb_kernel<<<(L_WIN * (DMODEL / 2) + 255) / 256, 256, 0, stream>>>(PE);
  gather_emb_kernel<<<(MROWS * 165 + 255) / 256, 256, 0, stream>>>(XE, AE);

  // token embedding GEMM + PE epilogue: [2048,165] x [165,512]
  gemm(stream, AE, TOKW, nullptr, PE, OUT, MROWS, DMODEL, 165,
       165, DMODEL, DMODEL, 0, 1.f, ACT_NONE, L_WIN, DMODEL);

  // d_out layout: out | series(3) | prior(3) | sigma(3)
  float* out_f = (float*)d_out;
  const long LSZ = (long)NB * NH * L_WIN * L_WIN;          // 8,388,608
  const long SER_BASE = (long)NB * L_WIN * CIN;            // 112,640
  const long PRI_BASE = SER_BASE + E_LAYERS * LSZ;
  const long SGO_BASE = PRI_BASE + E_LAYERS * LSZ;
  const long BLD = (long)L_WIN * DMODEL;                   // 262,144

  for (int ly = 0; ly < E_LAYERS; ++ly) {
    const int pb = 23 + ly * 18;
    const float* wq   = (const float*)d_in[pb + 0];
    const float* bq   = (const float*)d_in[pb + 1];
    const float* wk   = (const float*)d_in[pb + 2];
    const float* bk   = (const float*)d_in[pb + 3];
    const float* wv   = (const float*)d_in[pb + 4];
    const float* bv   = (const float*)d_in[pb + 5];
    const float* wsig = (const float*)d_in[pb + 6];
    const float* bsig = (const float*)d_in[pb + 7];
    const float* wo   = (const float*)d_in[pb + 8];
    const float* bo   = (const float*)d_in[pb + 9];
    const float* c1w  = (const float*)d_in[pb + 10];
    const float* c1b  = (const float*)d_in[pb + 11];
    const float* c2w  = (const float*)d_in[pb + 12];
    const float* c2b  = (const float*)d_in[pb + 13];
    const float* ln1g = (const float*)d_in[pb + 14];
    const float* ln1b = (const float*)d_in[pb + 15];
    const float* ln2g = (const float*)d_in[pb + 16];
    const float* ln2b = (const float*)d_in[pb + 17];

    // QKV projections
    gemm(stream, OUT, wq, bq, nullptr, Q,  MROWS, DMODEL, DMODEL, DMODEL, DMODEL, DMODEL, 0, 1.f, ACT_NONE, 1, 1);
    gemm(stream, OUT, wk, bk, nullptr, Kb, MROWS, DMODEL, DMODEL, DMODEL, DMODEL, DMODEL, 0, 1.f, ACT_NONE, 1, 1);
    gemm(stream, OUT, wv, bv, nullptr, V,  MROWS, DMODEL, DMODEL, DMODEL, DMODEL, DMODEL, 0, 1.f, ACT_NONE, 1, 1);
    sigma_head_kernel<<<(MROWS * NH + 255) / 256, 256, 0, stream>>>(OUT, wsig, bsig, SIG);

    // scores = Q_h K_h^T * 1/sqrt(64), written straight into d_out series region
    float* ser = out_f + SER_BASE + (long)ly * LSZ;
    gemm(stream, Q, Kb, nullptr, nullptr, ser, L_WIN, L_WIN, DHEAD,
         DMODEL, DMODEL, L_WIN, /*bt=*/1, 0.125f, ACT_NONE, 1, 1,
         /*Z=*/NB * NH, /*batchH=*/NH,
         BLD, DHEAD, BLD, DHEAD, (long)NH * L_WIN * L_WIN, (long)L_WIN * L_WIN);

    // in-place softmax over last dim
    softmax_rows<<<(NB * NH * L_WIN + 7) / 8, 256, 0, stream>>>(ser, NB * NH * L_WIN);

    // prior + sigma broadcast (single fused pass)
    prior_sigma_kernel<<<(int)((LSZ + 255) / 256), 256, 0, stream>>>(
        SIG, out_f + PRI_BASE + (long)ly * LSZ, out_f + SGO_BASE + (long)ly * LSZ, LSZ);

    // new_x = series @ V_h
    gemm(stream, ser, V, nullptr, nullptr, ATT, L_WIN, DHEAD, L_WIN,
         L_WIN, DMODEL, DMODEL, /*bt=*/0, 1.f, ACT_NONE, 1, 1,
         /*Z=*/NB * NH, /*batchH=*/NH,
         (long)NH * L_WIN * L_WIN, (long)L_WIN * L_WIN, BLD, DHEAD, BLD, DHEAD);

    // output projection, residual LN
    gemm(stream, ATT, wo, bo, nullptr, TMP, MROWS, DMODEL, DMODEL, DMODEL, DMODEL, DMODEL, 0, 1.f, ACT_NONE, 1, 1);
    ln_residual<<<(MROWS + 7) / 8, 256, 0, stream>>>(OUT, TMP, ln1g, ln1b, XR, MROWS);

    // FFN with exact GELU, residual LN
    gemm(stream, XR, c1w, c1b, nullptr, FF,  MROWS, DMODEL, DMODEL, DMODEL, DMODEL, DMODEL, 0, 1.f, ACT_GELU, 1, 1);
    gemm(stream, FF, c2w, c2b, nullptr, TMP, MROWS, DMODEL, DMODEL, DMODEL, DMODEL, DMODEL, 0, 1.f, ACT_NONE, 1, 1);
    ln_residual<<<(MROWS + 7) / 8, 256, 0, stream>>>(XR, TMP, ln2g, ln2b, OUT, MROWS);
  }

  // final LN + projection to C_OUT=55, written to d_out head
  ln_residual<<<(MROWS + 7) / 8, 256, 0, stream>>>(OUT, nullptr, LNFG, LNFB, TMP, MROWS);
  gemm(stream, TMP, PROJW, PROJB, nullptr, out_f, MROWS, CIN, DMODEL,
       DMODEL, CIN, CIN, 0, 1.f, ACT_NONE, 1, 1);
}
